// UMT5Attention_55336358642400
// MI455X (gfx1250) — matrix operand order repaired and checked by reading, not compiled, not run on hardware
//
#include <hip/hip_runtime.h>
#include <math.h>

// ---------------------------------------------------------------------------
// UMT5 encoder self-attention, bf16 WMMA pipeline for gfx1250 (MI455X).
// Async global->LDS staging (ASYNCcnt DMA path) where no transpose is
// required; K block staged once per workgroup instead of 4x per wave.
// ---------------------------------------------------------------------------

typedef __attribute__((ext_vector_type(16))) __bf16 v16bf;
typedef __attribute__((ext_vector_type(8)))  float  v8f;
typedef __attribute__((ext_vector_type(4)))  int    v4i;

union BF16x16 { v16bf v; uint4 u[2]; };

__device__ __forceinline__ v8f wmma_bf16(v16bf a, v16bf b, v8f c) {
    return __builtin_amdgcn_wmma_f32_16x16x32_bf16(
        /*neg_a=*/false, a, /*neg_b=*/false, b,
        /*c_mod=*/(short)0, c, /*reuse_a=*/false, /*reuse_b=*/false);
}

// ---- gfx1250 async global->LDS (guarded; sync fallback keeps both toolchains
// ---- compiling). Probe-confirmed param0 = v4i in AS1 (global), global-first.
#if defined(__has_builtin)
#  if __has_builtin(__builtin_amdgcn_global_load_async_to_lds_b128)
#    define ASYNC_LDS 1
#  endif
#endif
#ifndef ASYNC_LDS
#  define ASYNC_LDS 0
#endif

typedef __attribute__((address_space(1))) v4i* gv4i_p;   // global int4*
typedef __attribute__((address_space(3))) v4i* lv4i_p;   // LDS int4*

__device__ __forceinline__ void stage16B(const void* g, void* l) {
#if ASYNC_LDS
    __builtin_amdgcn_global_load_async_to_lds_b128((gv4i_p)g, (lv4i_p)l, 0, 0);
#else
    *(uint4*)l = *(const uint4*)g;
#endif
}

__device__ __forceinline__ void stage_wait() {
#if ASYNC_LDS
#  if __has_builtin(__builtin_amdgcn_s_wait_asynccnt)
    __builtin_amdgcn_s_wait_asynccnt(0);
#  else
    asm volatile("s_wait_asynccnt 0x0" ::: "memory");
#  endif
#endif
}

// ---------------------------------------------------------------------------
// Stage 1: f32 -> bf16 conversion (vectorized x4)
// ---------------------------------------------------------------------------
__global__ void cvt_f32_to_bf16(const float* __restrict__ src,
                                __bf16* __restrict__ dst, int n4) {
    int i = blockIdx.x * blockDim.x + threadIdx.x;
    if (i >= n4) return;
    float4 f = ((const float4*)src)[i];
    union { __bf16 e[4]; uint2 u; } t;
    t.e[0] = (__bf16)f.x; t.e[1] = (__bf16)f.y;
    t.e[2] = (__bf16)f.z; t.e[3] = (__bf16)f.w;
    ((uint2*)dst)[i] = t.u;
}

// ---------------------------------------------------------------------------
// Stage 2: T5 relative position bias table  biasTab[h][delta+2047]
// ---------------------------------------------------------------------------
__global__ void bias_table_kernel(const float* __restrict__ rel_bias,
                                  float* __restrict__ tab) {
    const int NTAB = 16 * 4095;
    int i = blockIdx.x * blockDim.x + threadIdx.x;
    if (i >= NTAB) return;
    int h = i / 4095;
    int delta = (i % 4095) - 2047;          // key - query
    int b = (delta > 0) ? 16 : 0;           // num_buckets/2 = 16
    int rp = (delta < 0) ? -delta : delta;
    int v;
    if (rp < 8) {                           // max_exact = 8
        v = rp;
    } else {
        float lr = __logf((float)rp * 0.125f) / __logf(16.0f); // log(128/8)=log16
        int large = 8 + (int)(lr * 8.0f);
        v = (large < 15) ? large : 15;
    }
    tab[i] = rel_bias[(b + v) * 16 + h];
}

// ---------------------------------------------------------------------------
// Stage 3/5: bf16 WMMA GEMM, C = A(MxK) * B(KxN).  BM=128 BN=64 BK=32,
// 256 threads = 8 waves, each wave owns a 16x64 strip.
// A tile staged via async global->LDS DMA; B tile transposed manually.
// ---------------------------------------------------------------------------
template <bool OUT_F32>
__global__ __launch_bounds__(256) void gemm_bf16(const __bf16* __restrict__ A,
                                                 const __bf16* __restrict__ B,
                                                 void* __restrict__ Cout,
                                                 int M, int N, int K) {
    __shared__ __bf16 As[128][40];   // [m][k], stride 40 avoids bank conflicts
    __shared__ __bf16 Bs[64][40];    // transposed: [n][k]

    const int tid  = threadIdx.x;
    const int lane = tid & 31;
    const int wave = tid >> 5;
    const int mBase = blockIdx.x * 128;
    const int nBase = blockIdx.y * 64;

    const int l15   = lane & 15;
    const int abase = (lane < 16) ? 0 : 8;    // A half-K base / C row offset
    const int bbase = (lane < 16) ? 0 : 16;   // unused (B fragments use abase)

    const v8f zero = {0.f, 0.f, 0.f, 0.f, 0.f, 0.f, 0.f, 0.f};
    v8f acc[4] = {zero, zero, zero, zero};

    for (int kb = 0; kb < K; kb += 32) {
        __syncthreads();
        {   // stage A tile 128x32 via async DMA (16 contiguous bf16 / thread)
            int row = tid >> 1;
            int col = (tid & 1) * 16;
            const __bf16* src = A + (size_t)(mBase + row) * K + kb + col;
            stage16B(src,     &As[row][col]);
            stage16B(src + 8, &As[row][col + 8]);
        }
        {   // stage B tile 32x64, transposed into [n][k]
            int k  = tid >> 3;
            int n0 = (tid & 7) * 8;
            union { uint4 u; __bf16 e[8]; } t;
            t.u = *(const uint4*)(B + (size_t)(kb + k) * N + nBase + n0);
#pragma unroll
            for (int e = 0; e < 8; ++e) Bs[n0 + e][k] = t.e[e];
        }
        stage_wait();
        __syncthreads();

        BF16x16 afr;   // A fragment: row = l15, K halves {abase, abase+16}
        afr.u[0] = *(const uint4*)(&As[wave * 16 + l15][abase]);
        afr.u[1] = *(const uint4*)(&As[wave * 16 + l15][abase + 16]);
#pragma unroll
        for (int t4 = 0; t4 < 4; ++t4) {
            BF16x16 bfr;  // B fragment: n = t4*16+l15, K halves {abase, abase+16}
            bfr.u[0] = *(const uint4*)(&Bs[t4 * 16 + l15][abase]);
            bfr.u[1] = *(const uint4*)(&Bs[t4 * 16 + l15][abase + 16]);
            acc[t4] = wmma_bf16(afr.v, bfr.v, acc[t4]);
        }
    }

#pragma unroll
    for (int t4 = 0; t4 < 4; ++t4) {
        int nIdx = nBase + t4 * 16 + l15;
#pragma unroll
        for (int r = 0; r < 8; ++r) {
            int mIdx = mBase + wave * 16 + abase + r;
            if constexpr (OUT_F32)
                ((float*)Cout)[(size_t)mIdx * N + nIdx] = acc[t4][r];
            else
                ((__bf16*)Cout)[(size_t)mIdx * N + nIdx] = (__bf16)acc[t4][r];
        }
    }
}

// ---------------------------------------------------------------------------
// Stage 4: flash attention per (head, 64-query block). 4 waves x 16 rows.
// Key blocks of 32; online softmax; no score materialization.
// K block: async DMA to LDS, shared by all 4 waves (was 4x global re-read).
// V block: manual transposed staging (DMA cannot transpose).
// ---------------------------------------------------------------------------
__global__ __launch_bounds__(128) void attn_flash(const __bf16* __restrict__ Q,
                                                  const __bf16* __restrict__ Kc,
                                                  const __bf16* __restrict__ Vc,
                                                  const float* __restrict__ biasTab,
                                                  __bf16* __restrict__ ctx) {
    constexpr int S = 2048, HD = 64, LD = 1024;
    __shared__ __bf16 Ks[32][72];       // K block row-major: [key][d] (+pad)
    __shared__ __bf16 Vt[64][40];       // V block transposed: [d][key]
    __shared__ __bf16 Ps[4][16][40];    // per-wave P scratch: [m][k]

    const int tid  = threadIdx.x;
    const int lane = tid & 31;
    const int wave = tid >> 5;
    const int head  = blockIdx.y;
    const int qbase = blockIdx.x * 64 + wave * 16;

    const int l15   = lane & 15;
    const int abase = (lane < 16) ? 0 : 8;
    const int bbase = (lane < 16) ? 0 : 16;

    // Q fragments for this wave's 16 rows (held in registers all loop long)
    BF16x16 qa[2];
    {
        const __bf16* qrow = Q + (size_t)(qbase + l15) * LD + head * HD;
#pragma unroll
        for (int c = 0; c < 2; ++c) {
            qa[c].u[0] = *(const uint4*)(qrow + c * 32 + abase);
            qa[c].u[1] = *(const uint4*)(qrow + c * 32 + abase + 16);
        }
    }

    const v8f zero = {0.f, 0.f, 0.f, 0.f, 0.f, 0.f, 0.f, 0.f};
    v8f o[4] = {zero, zero, zero, zero};
    float mrow[8], lrow[8];
#pragma unroll
    for (int r = 0; r < 8; ++r) { mrow[r] = -1.0e30f; lrow[r] = 0.0f; }

    const float* btab = biasTab + head * 4095;

    for (int kb = 0; kb < S; kb += 32) {
        __syncthreads();
        {   // async stage of K block (32 keys x 64 dims), row-major
            int key = tid >> 2;            // 0..31
            int d0  = (tid & 3) * 16;      // 0,16,32,48
            const __bf16* src = Kc + (size_t)(kb + key) * LD + head * HD + d0;
            stage16B(src,     &Ks[key][d0]);
            stage16B(src + 8, &Ks[key][d0 + 8]);
        }
        {   // cooperative transposed stage of V block (32 keys x 64 dims)
            int key = tid >> 2;            // 0..31
            int d0  = (tid & 3) * 16;      // 0,16,32,48
            union { uint4 u[2]; __bf16 e[16]; } t;
            const uint4* src = (const uint4*)(Vc + (size_t)(kb + key) * LD + head * HD + d0);
            t.u[0] = src[0];
            t.u[1] = src[1];
#pragma unroll
            for (int e = 0; e < 16; ++e) Vt[d0 + e][key] = t.e[e];
        }
        stage_wait();
        __syncthreads();

        if (kb + 32 < S)   // gfx1250 speculative prefetch of next K block
            __builtin_prefetch(Kc + (size_t)(kb + 32 + l15) * LD + head * HD, 0, 1);

        // ----- scores: two 16x16 tiles (keys kb..kb+15, kb+16..kb+31) -----
        float sc[2][8];
#pragma unroll
        for (int st = 0; st < 2; ++st) {
            v8f s = zero;
            int key = kb + st * 16 + l15;           // this lane's column
            const __bf16* krow = &Ks[st * 16 + l15][0];  // LDS, row-major == n-major
#pragma unroll
            for (int c = 0; c < 2; ++c) {           // d-chunks 0..31, 32..63
                BF16x16 bfr;                        // B = K^T fragment
                bfr.u[0] = *(const uint4*)(krow + c * 32 + abase);
                bfr.u[1] = *(const uint4*)(krow + c * 32 + abase + 16);
                s = wmma_bf16(qa[c].v, bfr.v, s);
            }
#pragma unroll
            for (int r = 0; r < 8; ++r) {           // + relative position bias
                int qi = qbase + abase + r;
                sc[st][r] = s[r] + btab[key - qi + 2047];
            }
        }

        // ----- online softmax (row stats replicated across 16-lane groups) -----
#pragma unroll
        for (int r = 0; r < 8; ++r) {
            float v = fmaxf(sc[0][r], sc[1][r]);
            v = fmaxf(v, __shfl_xor(v, 1, 16));
            v = fmaxf(v, __shfl_xor(v, 2, 16));
            v = fmaxf(v, __shfl_xor(v, 4, 16));
            v = fmaxf(v, __shfl_xor(v, 8, 16));
            float mnew  = fmaxf(mrow[r], v);
            float scale = __expf(mrow[r] - mnew);
            mrow[r] = mnew;
            float p0 = __expf(sc[0][r] - mnew);
            float p1 = __expf(sc[1][r] - mnew);
            sc[0][r] = p0;
            sc[1][r] = p1;
            float rs = p0 + p1;
            rs += __shfl_xor(rs, 1, 16);
            rs += __shfl_xor(rs, 2, 16);
            rs += __shfl_xor(rs, 4, 16);
            rs += __shfl_xor(rs, 8, 16);
            lrow[r] = lrow[r] * scale + rs;
#pragma unroll
            for (int t4 = 0; t4 < 4; ++t4) o[t4][r] *= scale;
        }

        // ----- P: C-layout -> A-layout via per-wave LDS round trip -----
#pragma unroll
        for (int r = 0; r < 8; ++r) {
            Ps[wave][abase + r][l15]      = (__bf16)sc[0][r];
            Ps[wave][abase + r][16 + l15] = (__bf16)sc[1][r];
        }
        asm volatile("s_wait_dscnt 0x0" ::: "memory");  // wave-local LDS RAW fence
        BF16x16 pa;
        pa.u[0] = *(const uint4*)(&Ps[wave][l15][abase]);
        pa.u[1] = *(const uint4*)(&Ps[wave][l15][abase + 16]);

        // ----- O += P(16x32) x V(32x64) : 4 WMMAs -----
#pragma unroll
        for (int t4 = 0; t4 < 4; ++t4) {
            BF16x16 vfr;  // B fragment from transposed V: n = t4*16+l15
            vfr.u[0] = *(const uint4*)(&Vt[t4 * 16 + l15][abase]);
            vfr.u[1] = *(const uint4*)(&Vt[t4 * 16 + l15][abase + 16]);
            o[t4] = wmma_bf16(pa.v, vfr.v, o[t4]);
        }
    }

    // ----- epilogue: normalize and store ctx (bf16, [s][h*64+d]) -----
#pragma unroll
    for (int r = 0; r < 8; ++r) {
        float inv = 1.0f / lrow[r];
        int mIdx = qbase + abase + r;
#pragma unroll
        for (int t4 = 0; t4 < 4; ++t4)
            ctx[(size_t)mIdx * LD + head * HD + t4 * 16 + l15] =
                (__bf16)(o[t4][r] * inv);
    }
}

// ---------------------------------------------------------------------------
// Host orchestration
// ---------------------------------------------------------------------------
extern "C" void kernel_launch(void* const* d_in, const int* in_sizes, int n_in,
                              void* d_out, int out_size, void* d_ws, size_t ws_size,
                              hipStream_t stream) {
    (void)in_sizes; (void)n_in; (void)out_size; (void)ws_size;
    constexpr int S = 2048, D = 1024;

    const float* hs = (const float*)d_in[0];
    const float* Wq = (const float*)d_in[1];
    const float* Wk = (const float*)d_in[2];
    const float* Wv = (const float*)d_in[3];
    const float* Wo = (const float*)d_in[4];
    const float* rb = (const float*)d_in[5];

    char* ws = (char*)d_ws;
    size_t off = 0;
    auto alloc = [&](size_t bytes) -> void* {
        void* p = ws + off;
        off += (bytes + 255) & ~(size_t)255;
        return p;
    };
    __bf16* hs_bf  = (__bf16*)alloc((size_t)S * D * 2);
    __bf16* wq_bf  = (__bf16*)alloc((size_t)D * D * 2);
    __bf16* wk_bf  = (__bf16*)alloc((size_t)D * D * 2);
    __bf16* wv_bf  = (__bf16*)alloc((size_t)D * D * 2);
    __bf16* wo_bf  = (__bf16*)alloc((size_t)D * D * 2);
    __bf16* q_bf   = (__bf16*)alloc((size_t)S * D * 2);
    __bf16* k_bf   = (__bf16*)alloc((size_t)S * D * 2);
    __bf16* v_bf   = (__bf16*)alloc((size_t)S * D * 2);
    __bf16* ctx_bf = (__bf16*)alloc((size_t)S * D * 2);
    float*  btab   = (float*)alloc((size_t)16 * 4095 * 4);

    // stage 1: conversions
    {
        int n4 = S * D / 4;
        cvt_f32_to_bf16<<<(n4 + 255) / 256, 256, 0, stream>>>(hs, hs_bf, n4);
        n4 = D * D / 4;
        cvt_f32_to_bf16<<<(n4 + 255) / 256, 256, 0, stream>>>(Wq, wq_bf, n4);
        cvt_f32_to_bf16<<<(n4 + 255) / 256, 256, 0, stream>>>(Wk, wk_bf, n4);
        cvt_f32_to_bf16<<<(n4 + 255) / 256, 256, 0, stream>>>(Wv, wv_bf, n4);
        cvt_f32_to_bf16<<<(n4 + 255) / 256, 256, 0, stream>>>(Wo, wo_bf, n4);
    }
    // stage 2: bias table
    bias_table_kernel<<<(16 * 4095 + 255) / 256, 256, 0, stream>>>(rb, btab);

    // stage 3: Q/K/V projections
    dim3 gg(S / 128, D / 64);
    gemm_bf16<false><<<gg, 256, 0, stream>>>(hs_bf, wq_bf, q_bf, S, D, D);
    gemm_bf16<false><<<gg, 256, 0, stream>>>(hs_bf, wk_bf, k_bf, S, D, D);
    gemm_bf16<false><<<gg, 256, 0, stream>>>(hs_bf, wv_bf, v_bf, S, D, D);

    // stage 4: flash attention (32 query blocks x 16 heads)
    dim3 ga(S / 64, 16);
    attn_flash<<<ga, 128, 0, stream>>>(q_bf, k_bf, v_bf, btab, ctx_bf);

    // stage 5: output projection, f32 result
    gemm_bf16<true><<<gg, 256, 0, stream>>>(ctx_bf, wo_bf, d_out, S, D, D);
}
